// PicoDetHead_16088947491355
// MI455X (gfx1250) — compile-verified, matrix operand order checked by reading, not run on hardware
//
#include <hip/hip_runtime.h>
#include <math.h>

// ---------------------------------------------------------------------------
// PicoDet-style detection head, fully fused per 16x16 spatial tile.
// Matmuls (ghost 1x1 convs + cls/obj/reg heads) run on CDNA5 WMMA
// (v_wmma_f32_16x16x32_f16, f32 accumulate). Depthwise 3x3 / DFL / sigmoid
// epilogue on VALU. One HBM read of features, one HBM write of outputs.
// ---------------------------------------------------------------------------

typedef _Float16 half_t;
typedef __attribute__((ext_vector_type(16))) _Float16 v16h;
typedef __attribute__((ext_vector_type(8)))  _Float16 v8h;
typedef __attribute__((ext_vector_type(8)))  float    v8f;

#define NCLS 80
#define A_TOTAL 8400
#define NBATCH 16

// LDS layout (bytes). FSTR/WSTR pad the 96-wide rows to 104 halves (208 B)
// so strided WMMA-fragment row accesses spread across the 64 LDS banks.
#define FSTR 104   // halves per sFeat / sW row
#define OFF_FEAT 0        // 400 * 104 * 2 = 83200   (20x20 x 96ch f16)
#define OFF_Y1   83200    // 400 *  48 * 2 = 38400   (20x20 x 48 f16)
#define OFF_Z1   121600   // 324 *  48 * 2 = 31104   (18x18 x 48 f16)
#define OFF_Y2   152704   // 324 *  48 * 2 = 31104
#define OFF_Z2   183808   // 256 *  48 * 2 = 24576   (16x16 x 48 f16)
#define OFF_W    208384   //  96 * 104 * 2 = 19968   (weights [N][K] f16)
#define OFF_LC   228352   // 256 *  96 * 2 = 49152   (cls+obj logits f16)
#define OFF_LR   277504   // 256 *  32 * 2 = 16384   (reg logits f16)
#define OFF_P    293888   // 720 floats params       (dw, bn, biases)
#define LDS_TOTAL 296768  // ~290 KB < 320 KB/WGP

__device__ __forceinline__ float relu6f(float x) { return fminf(fmaxf(x, 0.f), 6.f); }

// Build a v16h WMMA fragment from two contiguous 16-byte LDS chunks.
__device__ __forceinline__ v16h frag2(const half_t* c0, const half_t* c1) {
  v8h lo = *(const v8h*)c0;
  v8h hi = *(const v8h*)c1;
  return __builtin_shufflevector(lo, hi, 0,1,2,3,4,5,6,7,8,9,10,11,12,13,14,15);
}

// One 16x16 output tile of C = A[16x96] * B[96x16], K split into 3 steps of 32.
// A channel k comes from arowY[k] when k<48 else arowZ[k] (arowZ pre-biased by
// -48), so concat(y, z) inputs need no copy. For single-buffer A pass
// arowZ == arowY. kb = per-lane K base (0 for lanes 0-15, 8 for lanes 16-31),
// matching the ISA 16-bit A/B VGPR layout; K(e) = kb + (e&7) + (e&8)*2.
__device__ __forceinline__ v8f mm96(const half_t* arowY, const half_t* arowZ,
                                    const half_t* brow, int kb) {
  v8f acc = {};
#pragma unroll
  for (int kt = 0; kt < 3; ++kt) {
    const int k0 = kt * 32 + kb;
    const int k1 = k0 + 16;
    const half_t* a0 = (k0 < 48) ? (arowY + k0) : (arowZ + k0);
    const half_t* a1 = (k1 < 48) ? (arowY + k1) : (arowZ + k1);
    v16h a = frag2(a0, a1);
    v16h b = frag2(brow + k0, brow + k1);
    acc = __builtin_amdgcn_wmma_f32_16x16x32_f16(false, a, false, b,
                                                 (short)0, acc, false, false);
  }
  return acc;
}

extern "C" __global__ void __launch_bounds__(256)
picodet_head_level(const float* __restrict__ feat, int H, int W, int tilesX,
                   float strd, int aBase, int level,
                   const float* __restrict__ primary_w,
                   const float* __restrict__ bn1_s, const float* __restrict__ bn1_b,
                   const float* __restrict__ cheap_w,
                   const float* __restrict__ bn2_s, const float* __restrict__ bn2_b,
                   const float* __restrict__ cls_w, const float* __restrict__ cls_b,
                   const float* __restrict__ obj_w, const float* __restrict__ obj_b,
                   const float* __restrict__ reg_w, const float* __restrict__ reg_b,
                   float* __restrict__ out) {
  extern __shared__ char smem[];
  half_t* sFeat = (half_t*)(smem + OFF_FEAT);
  half_t* sY1   = (half_t*)(smem + OFF_Y1);
  half_t* sZ1   = (half_t*)(smem + OFF_Z1);
  half_t* sY2   = (half_t*)(smem + OFF_Y2);
  half_t* sZ2   = (half_t*)(smem + OFF_Z2);
  half_t* sW    = (half_t*)(smem + OFF_W);
  half_t* sLC   = (half_t*)(smem + OFF_LC);
  half_t* sLR   = (half_t*)(smem + OFF_LR);
  float*  sP    = (float*)(smem + OFF_P);
  float*  sDW   = sP;          // 432 (48ch x 3x3)
  float*  sB1s  = sP + 432;    // 48
  float*  sB1b  = sP + 480;
  float*  sB2s  = sP + 528;
  float*  sB2b  = sP + 576;
  float*  sHB   = sP + 624;    // 96 (head bias, padded)

  const int tid  = threadIdx.x;
  const int lane = tid & 31;
  const int wave = tid >> 5;
  const int nl   = lane & 15;            // N/M within tile
  const int kb   = (lane & 16) ? 8 : 0;  // per-lane K base for A/B frags
  const int mb   = (lane & 16) ? 8 : 0;  // per-lane M base for C/D frags

  const int bIdx = blockIdx.y;
  const int tX = blockIdx.x % tilesX, tY = blockIdx.x / tilesX;
  const int tx0 = tX * 16, ty0 = tY * 16;

  // ---- Stage 0: load 20x20x96 feature tile as f16 with zero padding -------
  const float* fbase = feat + (size_t)bIdx * 96 * H * W;
  for (int idx = tid; idx < 400 * 96; idx += 256) {
    const int c = idx / 400, p = idx % 400;
    const int iy = p / 20, ix = p % 20;
    const int gy = ty0 - 2 + iy, gx = tx0 - 2 + ix;
    float v = 0.f;
    if ((unsigned)gy < (unsigned)H && (unsigned)gx < (unsigned)W)
      v = fbase[((size_t)c * H + gy) * W + gx];
    sFeat[p * FSTR + c] = (half_t)v;
  }

  for (int br = 0; br < 2; ++br) {       // 0 = cls tower, 1 = reg tower
    for (int li = 0; li < 2; ++li) {     // ghost layers
      // ---- stage weights/params for this (branch, layer) ------------------
      const int wo = (br * 2 + li);
      __syncthreads();  // previous readers of sW/sP done
      const float* pw = primary_w + (size_t)wo * 48 * 96;
      for (int i = tid; i < 48 * 96; i += 256)
        sW[(i / 96) * FSTR + (i % 96)] = (half_t)pw[i];
      const float* dw = cheap_w + (size_t)wo * 432;
      for (int i = tid; i < 432; i += 256) sDW[i] = dw[i];
      for (int i = tid; i < 48; i += 256) {
        sB1s[i] = bn1_s[wo * 48 + i]; sB1b[i] = bn1_b[wo * 48 + i];
        sB2s[i] = bn2_s[wo * 48 + i]; sB2b[i] = bn2_b[wo * 48 + i];
      }
      __syncthreads();

      if (li == 0) {
        // ---- primary 1x1: [400px x 96] x [96 x 48] -> sY1 (20x20) ---------
        for (int t = wave; t < 25 * 3; t += 8) {
          const int mt = t / 3, nt = t % 3;
          const half_t* arow = sFeat + (mt * 16 + nl) * FSTR;
          const half_t* brow = sW + (nt * 16 + nl) * FSTR;
          v8f acc = mm96(arow, arow, brow, kb);
          const int n = nt * 16 + nl;
          const float s1 = sB1s[n], b1 = sB1b[n];
#pragma unroll
          for (int r = 0; r < 8; ++r) {
            const int p = mt * 16 + mb + r;
            const int iy = p / 20, ix = p % 20;
            const int gy = ty0 - 2 + iy, gx = tx0 - 2 + ix;
            const bool in = (unsigned)gy < (unsigned)H && (unsigned)gx < (unsigned)W;
            // zero outside image => correct SAME zero padding for depthwise
            sY1[p * 48 + n] = (half_t)(in ? relu6f(acc[r] * s1 + b1) : 0.f);
          }
        }
        __syncthreads();
        // ---- depthwise 3x3: sY1(20x20) -> sZ1(18x18) ----------------------
        for (int idx = tid; idx < 324 * 48; idx += 256) {
          const int p = idx / 48, c = idx % 48;
          const int iy = p / 18, ix = p % 18;
          const half_t* base = sY1 + (iy * 20 + ix) * 48 + c;
          float a = 0.f;
#pragma unroll
          for (int dy = 0; dy < 3; ++dy)
#pragma unroll
            for (int dx = 0; dx < 3; ++dx)
              a += (float)base[(dy * 20 + dx) * 48] * sDW[c * 9 + dy * 3 + dx];
          sZ1[idx] = (half_t)relu6f(a * sB2s[c] + sB2b[c]);
        }
        __syncthreads();
      } else {
        // ---- primary 1x1 on concat(y1,z1): [324px x 96] x [96x48] -> sY2 --
        for (int t = wave; t < 21 * 3; t += 8) {
          const int mt = t / 3, nt = t % 3;
          int am = mt * 16 + nl; if (am > 323) am = 323;  // pad tile 21
          const int aiy = am / 18, aix = am % 18;
          const half_t* arowY = sY1 + ((aiy + 1) * 20 + aix + 1) * 48;
          const half_t* arowZ = sZ1 + am * 48 - 48;  // +k for k>=48
          const half_t* brow = sW + (nt * 16 + nl) * FSTR;
          v8f acc = mm96(arowY, arowZ, brow, kb);
          const int n = nt * 16 + nl;
          const float s1 = sB1s[n], b1 = sB1b[n];
#pragma unroll
          for (int r = 0; r < 8; ++r) {
            const int p = mt * 16 + mb + r;
            if (p < 324) {
              const int iy = p / 18, ix = p % 18;
              const int gy = ty0 - 1 + iy, gx = tx0 - 1 + ix;
              const bool in = (unsigned)gy < (unsigned)H && (unsigned)gx < (unsigned)W;
              sY2[p * 48 + n] = (half_t)(in ? relu6f(acc[r] * s1 + b1) : 0.f);
            }
          }
        }
        __syncthreads();
        // ---- depthwise 3x3: sY2(18x18) -> sZ2(16x16) ----------------------
        for (int idx = tid; idx < 256 * 48; idx += 256) {
          const int p = idx / 48, c = idx % 48;
          const int iy = p >> 4, ix = p & 15;
          const half_t* base = sY2 + (iy * 18 + ix) * 48 + c;
          float a = 0.f;
#pragma unroll
          for (int dy = 0; dy < 3; ++dy)
#pragma unroll
            for (int dx = 0; dx < 3; ++dx)
              a += (float)base[(dy * 18 + dx) * 48] * sDW[c * 9 + dy * 3 + dx];
          sZ2[idx] = (half_t)relu6f(a * sB2s[c] + sB2b[c]);
        }
        __syncthreads();
      }
    }

    // ---- head matmul for this branch on x2 = concat(y2, z2), 256 px -------
    if (br == 0) {
      // cls (80) + obj (1), padded to N=96
      const float* cw = cls_w + (size_t)level * NCLS * 96;
      const float* ow = obj_w + (size_t)level * 96;
      for (int i = tid; i < 96 * 96; i += 256) {
        const int n = i / 96, k = i % 96;
        float v = 0.f;
        if (n < NCLS) v = cw[n * 96 + k];
        else if (n == NCLS) v = ow[k];
        sW[n * FSTR + k] = (half_t)v;
      }
      for (int i = tid; i < 96; i += 256) {
        float v = 0.f;
        if (i < NCLS) v = cls_b[level * NCLS + i];
        else if (i == NCLS) v = obj_b[level];
        sHB[i] = v;
      }
      __syncthreads();
      for (int t = wave; t < 16 * 6; t += 8) {
        const int mt = t / 6, nt = t % 6;
        const int am = mt * 16 + nl;
        const int aiy = am >> 4, aix = am & 15;
        const half_t* arowY = sY2 + ((aiy + 1) * 18 + aix + 1) * 48;
        const half_t* arowZ = sZ2 + am * 48 - 48;
        const half_t* brow = sW + (nt * 16 + nl) * FSTR;
        v8f acc = mm96(arowY, arowZ, brow, kb);
        const int n = nt * 16 + nl;
        const float bias = sHB[n];
#pragma unroll
        for (int r = 0; r < 8; ++r)
          sLC[(mt * 16 + mb + r) * 96 + n] = (half_t)(acc[r] + bias);
      }
      __syncthreads();
    } else {
      const float* rw = reg_w + (size_t)level * 32 * 96;
      for (int i = tid; i < 32 * 96; i += 256)
        sW[(i / 96) * FSTR + (i % 96)] = (half_t)rw[i];
      for (int i = tid; i < 32; i += 256) sHB[i] = reg_b[level * 32 + i];
      __syncthreads();
      for (int t = wave; t < 16 * 2; t += 8) {
        const int mt = t / 2, nt = t % 2;
        const int am = mt * 16 + nl;
        const int aiy = am >> 4, aix = am & 15;
        const half_t* arowY = sY2 + ((aiy + 1) * 18 + aix + 1) * 48;
        const half_t* arowZ = sZ2 + am * 48 - 48;
        const half_t* brow = sW + (nt * 16 + nl) * FSTR;
        v8f acc = mm96(arowY, arowZ, brow, kb);
        const int n = nt * 16 + nl;
        const float bias = sHB[n];
#pragma unroll
        for (int r = 0; r < 8; ++r)
          sLR[(mt * 16 + mb + r) * 32 + n] = (half_t)(acc[r] + bias);
      }
      __syncthreads();
    }
  }

  // ---- epilogue: DFL decode + anchors -> boxes; sigmoid(cls+obj) -> scores
  {
    const int p = tid;                    // one pixel per thread
    const int iy = p >> 4, ix = p & 15;
    const int gy = ty0 + iy, gx = tx0 + ix;
    if (gy < H && gx < W) {
      float l[32];
#pragma unroll
      for (int i = 0; i < 32; ++i) l[i] = (float)sLR[p * 32 + i];
      float ltrb[4];
#pragma unroll
      for (int j = 0; j < 4; ++j) {
        float m = l[j * 8];
#pragma unroll
        for (int r = 1; r < 8; ++r) m = fmaxf(m, l[j * 8 + r]);
        float sum = 0.f, num = 0.f;
#pragma unroll
        for (int r = 0; r < 8; ++r) {
          const float e = __expf(l[j * 8 + r] - m);
          sum += e; num += e * (float)r;
        }
        ltrb[j] = (num / sum) * strd;
      }
      const float ax = ((float)gx + 0.5f) * strd;
      const float ay = ((float)gy + 0.5f) * strd;
      const size_t a = (size_t)bIdx * A_TOTAL + aBase + (size_t)gy * W + gx;
      float* bx = out + a * 4;
      bx[0] = ax - ltrb[0]; bx[1] = ay - ltrb[1];
      bx[2] = ax + ltrb[2]; bx[3] = ay + ltrb[3];
      const float obj = (float)sLC[p * 96 + NCLS];
      float* sc = out + (size_t)NBATCH * A_TOTAL * 4 + a * NCLS;
      for (int c = 0; c < NCLS; ++c) {
        const float z = (float)sLC[p * 96 + c] + obj;
        sc[c] = 1.f / (1.f + __expf(-z));
      }
    }
  }
}

extern "C" void kernel_launch(void* const* d_in, const int* in_sizes, int n_in,
                              void* d_out, int out_size, void* d_ws, size_t ws_size,
                              hipStream_t stream) {
  (void)in_sizes; (void)n_in; (void)d_ws; (void)ws_size; (void)out_size;
  const float* feat0     = (const float*)d_in[0];
  const float* feat1     = (const float*)d_in[1];
  const float* feat2     = (const float*)d_in[2];
  const float* primary_w = (const float*)d_in[3];
  const float* bn1_s     = (const float*)d_in[4];
  const float* bn1_b     = (const float*)d_in[5];
  const float* cheap_w   = (const float*)d_in[6];
  const float* bn2_s     = (const float*)d_in[7];
  const float* bn2_b     = (const float*)d_in[8];
  const float* cls_w     = (const float*)d_in[9];
  const float* cls_b     = (const float*)d_in[10];
  const float* obj_w     = (const float*)d_in[11];
  const float* obj_b     = (const float*)d_in[12];
  const float* reg_w     = (const float*)d_in[13];
  const float* reg_b     = (const float*)d_in[14];
  float* out = (float*)d_out;

  // level 0: 80x80, stride 8, 5x5 tiles; level 1: 40x40, 3x3; level 2: 20x20, 2x2
  picodet_head_level<<<dim3(25, NBATCH), 256, LDS_TOTAL, stream>>>(
      feat0, 80, 80, 5, 8.f, 0, 0, primary_w, bn1_s, bn1_b, cheap_w, bn2_s,
      bn2_b, cls_w, cls_b, obj_w, obj_b, reg_w, reg_b, out);
  picodet_head_level<<<dim3(9, NBATCH), 256, LDS_TOTAL, stream>>>(
      feat1, 40, 40, 3, 16.f, 6400, 1, primary_w, bn1_s, bn1_b, cheap_w, bn2_s,
      bn2_b, cls_w, cls_b, obj_w, obj_b, reg_w, reg_b, out);
  picodet_head_level<<<dim3(4, NBATCH), 256, LDS_TOTAL, stream>>>(
      feat2, 20, 20, 2, 32.f, 8000, 2, primary_w, bn1_s, bn1_b, cheap_w, bn2_s,
      bn2_b, cls_w, cls_b, obj_w, obj_b, reg_w, reg_b, out);
}
